// CapLayer_90056874263182
// MI455X (gfx1250) — matrix operand order repaired and checked by reading, not compiled, not run on hardware
//
#include <hip/hip_runtime.h>
#include <hip/hip_bf16.h>
#include <math.h>

// Capsule routing layer, fully fused: one workgroup per batch element.
// pred (4608x16) and c/logits (16x4608) live in the 320KB WGP LDS; routing
// logits b[o,i] are represented implicitly as b0 + pred . Vacc^T (rank-16).
// All three GEMMs (pred, logits, s) run on WMMA:
//   pred   : v_wmma_f32_16x16x4_f32  (exact f32, K=8 as 2 steps)
//   logits : v_wmma_f32_16x16x32_f16 (K=16 zero-padded via zero B rows)
//   s      : v_wmma_f32_16x16x32_f16 (K accumulated over NI=4608)
// HBM traffic ~= read x once (75.5MB) -> ~3.3us roofline at 23.3 TB/s.

#define NUM_SHARED 32
#define IN_DIM     8
#define OUT_DIM    16
#define NUM_OUT    10
#define ROUTE      3
#define HW         144
#define NI         (NUM_SHARED * HW)   // 4608
#define NIP        (NI + 8)            // padded LDS row to break bank aliasing
#define BATCH      512
#define THREADS    256                 // 8 waves (wave32)

typedef __attribute__((ext_vector_type(16))) _Float16 v16h;
typedef __attribute__((ext_vector_type(8)))  float    v8f;
typedef __attribute__((ext_vector_type(2)))  float    v2f;

union AFrag16 { v16h v; unsigned u[8]; };

__global__ __launch_bounds__(THREADS)
void caps_routing_kernel(const float* __restrict__ x,
                         const float* __restrict__ W,
                         const float* __restrict__ Wb,
                         const float* __restrict__ b0,
                         float* __restrict__ out)
{
    // ---- LDS: ~307 KB of the 320 KB WGP budget ----
    __shared__ _Float16 predT[OUT_DIM][NIP];   // pred transposed: [d][i], f16
    __shared__ _Float16 ch[16][NIP];           // logits, then c: [o(pad16)][i]
    __shared__ _Float16 VaccHT[32][16];        // Vacc^T f16: [d(pad32)][o]
    __shared__ float    sPart[8][16][17];      // per-wave WMMA partials
    __shared__ float    VaccL[16][16];         // f32 master copy of Vacc
    __shared__ float    sFull[16][16];
    __shared__ float    coeff[16];

    const int b    = blockIdx.x;
    const int tid  = threadIdx.x;
    const int wave = tid >> 5;
    const int lane = tid & 31;
    const int m    = lane & 15;   // row/col index within 16
    const int h    = lane >> 4;   // lane half (WMMA K-half selector)

    const float* xb = x + (size_t)b * (NI * IN_DIM);

    // Init Vacc = 0 in both precisions (rows/cols >= NUM_OUT stay zero forever).
    if (tid < 256) (&VaccL[0][0])[tid] = 0.f;
    if (tid < 256) { (&VaccHT[0][0])[tid]       = (_Float16)0.f;
                     (&VaccHT[0][0])[tid + 256] = (_Float16)0.f; }

    // -------- Phase A: pred[i, o] = U[i,:] . W[s]^T + Wb[s]  (s = i/144) ----
#if __has_builtin(__builtin_amdgcn_wmma_f32_16x16x4_f32)
    for (int t = wave; t < NI / 16; t += 8) {
        const int s = t / (HW / 16);           // 9 tiles per shared group
        v8f acc = {};
        #pragma unroll
        for (int kk = 0; kk < 2; ++kk) {       // K=8 as two K=4 WMMA steps
            const int k0 = kk * 4 + 2 * h;
            v2f a  = *(const v2f*)(xb + (size_t)(t * 16 + m) * IN_DIM + k0);
            v2f bb = *(const v2f*)(W + (size_t)(s * OUT_DIM + m) * IN_DIM + k0);
            acc = __builtin_amdgcn_wmma_f32_16x16x4_f32(
                      false, a, false, bb, (short)0, acc, false, false);
        }
        const float bias = Wb[s * OUT_DIM + m];   // N (= o) index is m
        #pragma unroll
        for (int r = 0; r < 8; ++r)               // D row = r + 8h, col = m
            predT[m][t * 16 + 8 * h + r] = (_Float16)(acc[r] + bias);
    }
#else
    for (int gi = tid; gi < NI; gi += THREADS) {
        const int s = gi / HW;
        float u[IN_DIM];
        #pragma unroll
        for (int i = 0; i < IN_DIM; ++i) u[i] = xb[(size_t)gi * IN_DIM + i];
        for (int o = 0; o < OUT_DIM; ++o) {
            float acc = Wb[s * OUT_DIM + o];
            #pragma unroll
            for (int i = 0; i < IN_DIM; ++i)
                acc += u[i] * W[(s * OUT_DIM + o) * IN_DIM + i];
            predT[o][gi] = (_Float16)acc;
        }
    }
#endif
    __syncthreads();

    // -------- Phase B: 3 routing iterations, all on-chip --------
    for (int it = 0; it < ROUTE; ++it) {
        // B0: logits L[i,o] = pred[i,:].Vacc[o,:] via WMMA, written into ch.
        // B-fragment (Vacc^T) is tile-invariant: build once per iteration.
        // K=32 with d=16: B rows 16..31 are zero -> A VGPRs 4..7 can be zero.
        AFrag16 vfrag;
        #pragma unroll
        for (int r = 0; r < 8; ++r) {
            unsigned v = 0;
            if (r < 4) {
                unsigned lo = *(const unsigned short*)&VaccHT[8 * h + 2 * r][m];
                unsigned hi = *(const unsigned short*)&VaccHT[8 * h + 2 * r + 1][m];
                v = lo | (hi << 16);
            }
            vfrag.u[r] = v;   // r>=4 covers K=16..31 -> zero rows
        }
        for (int t = wave; t < NI / 16; t += 8) {
            const int i0 = t * 16;
            AFrag16 af;
            #pragma unroll
            for (int r = 0; r < 4; ++r) {     // K = 8h + 2r (+1): all < 16
                const int ka = 8 * h + 2 * r;
                unsigned lo = *(const unsigned short*)&predT[ka][i0 + m];
                unsigned hi = *(const unsigned short*)&predT[ka + 1][i0 + m];
                af.u[r] = lo | (hi << 16);
            }
            #pragma unroll
            for (int r = 4; r < 8; ++r) af.u[r] = 0;   // K>=16: B rows zero
            v8f acc = {};
            acc = __builtin_amdgcn_wmma_f32_16x16x32_f16(
                      false, af.v, false, vfrag.v, (short)0, acc, false, false);
            #pragma unroll
            for (int r = 0; r < 8; ++r)       // D: row i = r+8h, col o = m
                ch[m][i0 + 8 * h + r] = (_Float16)acc[r];
            // cols o=10..15 get exact zeros (Vacc cols are zero) -> keeps the
            // padded A-rows of the s-GEMM valid with no extra init.
        }
        __syncthreads();

        // B1: softmax over o per column i (logits read from ch, c written back).
        for (int i = tid; i < NI; i += THREADS) {
            float l[NUM_OUT];
            float mx = -1e30f;
            #pragma unroll
            for (int o = 0; o < NUM_OUT; ++o) {
                float v = (float)ch[o][i] + b0[(size_t)o * NI + i];
                l[o] = v;
                mx = fmaxf(mx, v);
            }
            float sum = 0.f;
            #pragma unroll
            for (int o = 0; o < NUM_OUT; ++o) { l[o] = __expf(l[o] - mx); sum += l[o]; }
            const float inv = 1.f / sum;
            #pragma unroll
            for (int o = 0; o < NUM_OUT; ++o) ch[o][i] = (_Float16)(l[o] * inv);
        }
        __syncthreads();

        // B2: s[16x16] = C[16 x NI] . pred[NI x 16] via v_wmma_f32_16x16x32_f16.
        v8f acc = {};
        for (int c4 = wave; c4 < NI / 32; c4 += 8) {
            const int i0 = c4 * 32;
            AFrag16 af, bf;
            #pragma unroll
            for (int r = 0; r < 8; ++r) {
                const int ka = ((r < 4) ? 0 : 16) + 8 * h + 2 * (r & 3);
                af.u[r] = *(const unsigned*)&ch[m][i0 + ka];
                const int kb = ((r < 4) ? 0 : 16) + 8 * h + 2 * (r & 3);
                bf.u[r] = *(const unsigned*)&predT[m][i0 + kb];
            }
            acc = __builtin_amdgcn_wmma_f32_16x16x32_f16(
                      false, af.v, false, bf.v, (short)0, acc, false, false);
        }
        #pragma unroll
        for (int r = 0; r < 8; ++r)          // D: row o = r+8h, col d = m
            sPart[wave][8 * h + r][m] = acc[r];
        __syncthreads();

        // B3: cross-wave reduce, squash, accumulate Vacc (f32 + f16T), emit v.
        if (tid < NUM_OUT * 16) {
            const int o = tid >> 4, d = tid & 15;
            float s = 0.f;
            #pragma unroll
            for (int w = 0; w < 8; ++w) s += sPart[w][o][d];
            sFull[o][d] = s;
        }
        __syncthreads();
        if (tid < NUM_OUT) {
            float n2 = 0.f;
            #pragma unroll
            for (int d = 0; d < 16; ++d) { float v = sFull[tid][d]; n2 += v * v; }
            const float nrm = sqrtf(n2);
            coeff[tid] = nrm / (1.f + n2);   // (n^2/(1+n^2))/n
        }
        __syncthreads();
        if (tid < NUM_OUT * 16) {
            const int o = tid >> 4, d = tid & 15;
            const float v = sFull[o][d] * coeff[o];
            const float vn = VaccL[o][d] + v;
            VaccL[o][d]  = vn;
            VaccHT[d][o] = (_Float16)vn;
            if (it == ROUTE - 1)
                out[(size_t)b * (NUM_OUT * 16) + tid] = v;
        }
        __syncthreads();
    }
}

extern "C" void kernel_launch(void* const* d_in, const int* in_sizes, int n_in,
                              void* d_out, int out_size, void* d_ws, size_t ws_size,
                              hipStream_t stream) {
    const float* x  = (const float*)d_in[0];
    const float* W  = (const float*)d_in[1];
    const float* Wb = (const float*)d_in[2];
    const float* b0 = (const float*)d_in[3];
    (void)in_sizes; (void)n_in; (void)out_size; (void)d_ws; (void)ws_size;
    caps_routing_kernel<<<dim3(BATCH), dim3(THREADS), 0, stream>>>(
        x, W, Wb, b0, (float*)d_out);
}
